// GreedyFeatureInit_35631048687924
// MI455X (gfx1250) — matrix operand order repaired and checked by reading, not compiled, not run on hardware
//
#include <hip/hip_runtime.h>
#include <hip/hip_bf16.h>

// GreedyFeatureInit for MI455X (gfx1250, wave32).
// B=32, N=16384, D=256, n_slots=8. Output [B, n_slots, D] f32.
//
// Pipeline:
//   1) init_norms: one wave per row. saliency=||x||, invnorm=1/max(||x||,eps),
//      score=saliency, fnorm_f16 = f16(x/||x||)  (256 MB cache in ws).
//   2) per slot: select_slot (argmax per batch + write output + sel_norm f16),
//      then suppress_wmma (v_wmma_f32_16x16x32_f16 GEMV: score *= 1-clip(sim)).
//   Fallback suppress_f32 (no f16 cache) if ws_size is too small.

typedef _Float16 v16h __attribute__((ext_vector_type(16)));
typedef _Float16 v8h  __attribute__((ext_vector_type(8)));
typedef float    v8f  __attribute__((ext_vector_type(8)));

#define EPSN 1e-12f

static __device__ __forceinline__ v16h mk16(v8h lo, v8h hi) {
  v16h r;
#pragma unroll
  for (int j = 0; j < 8; ++j) { r[j] = lo[j]; r[j + 8] = hi[j]; }
  return r;
}

// ---------------------------------------------------------------------------
// Pass 0: per-row norm, invnorm, score init, f16 normalized copy.
// One wave (32 lanes) per row of D=256 floats; lane handles 8 consecutive.
// ---------------------------------------------------------------------------
__global__ __launch_bounds__(256)
void init_norms(const float* __restrict__ feat,
                float* __restrict__ score,
                float* __restrict__ invn,
                _Float16* __restrict__ fn16,   // may be null (fallback path)
                long rowsTotal, int D) {
  long wave = (long)((blockIdx.x * blockDim.x + threadIdx.x) >> 5);
  int lane = threadIdx.x & 31;
  if (wave >= rowsTotal) return;

  const float* rp = feat + (size_t)wave * D + lane * 8;
  float x0 = rp[0], x1 = rp[1], x2 = rp[2], x3 = rp[3];
  float x4 = rp[4], x5 = rp[5], x6 = rp[6], x7 = rp[7];
  float ss = x0*x0 + x1*x1 + x2*x2 + x3*x3 + x4*x4 + x5*x5 + x6*x6 + x7*x7;
#pragma unroll
  for (int off = 16; off > 0; off >>= 1) ss += __shfl_xor(ss, off, 32);

  float nrm = sqrtf(ss);
  float inv = 1.0f / fmaxf(nrm, EPSN);
  if (lane == 0) { score[wave] = nrm; invn[wave] = inv; }

  if (fn16 != nullptr) {
    v8h h;
    h[0] = (_Float16)(x0 * inv); h[1] = (_Float16)(x1 * inv);
    h[2] = (_Float16)(x2 * inv); h[3] = (_Float16)(x3 * inv);
    h[4] = (_Float16)(x4 * inv); h[5] = (_Float16)(x5 * inv);
    h[6] = (_Float16)(x6 * inv); h[7] = (_Float16)(x7 * inv);
    *(v8h*)(fn16 + (size_t)wave * D + lane * 8) = h;
  }
}

// ---------------------------------------------------------------------------
// Per-slot selection: one block per batch. Argmax(score) over N (first-index
// tie-break like jnp.argmax), write selected row to d_out, write normalized
// selected row (f16) to selnorm, zero the picked score entry.
// blockDim.x == 256 == D.
// ---------------------------------------------------------------------------
__global__ __launch_bounds__(256)
void select_slot(const float* __restrict__ feat,
                 float* __restrict__ score,
                 _Float16* __restrict__ selnorm,   // [B, D]
                 float* __restrict__ out,          // [B, n_slots, D]
                 int s, int N, int D, int n_slots) {
  __shared__ float sv[256];
  __shared__ int   si[256];
  const int b = blockIdx.x;
  const int tid = threadIdx.x;

  const float* sc = score + (size_t)b * N;
  float best = -1.0f; int bidx = 0;
  for (int i = tid; i < N; i += blockDim.x) {
    float v = sc[i];
    if (v > best) { best = v; bidx = i; }   // strict '>' keeps first occurrence
  }
  sv[tid] = best; si[tid] = bidx;
  __syncthreads();
#pragma unroll
  for (int off = 128; off > 0; off >>= 1) {
    if (tid < off) {
      float v2 = sv[tid + off]; int i2 = si[tid + off];
      if (v2 > sv[tid] || (v2 == sv[tid] && i2 < si[tid])) { sv[tid] = v2; si[tid] = i2; }
    }
    __syncthreads();
  }
  const int idx = si[0];
  __syncthreads();

  // Load the selected row (D == blockDim), compute its norm.
  const float* sel = feat + ((size_t)b * N + idx) * D;
  float v = sel[tid];
  sv[tid] = v * v;
  __syncthreads();
#pragma unroll
  for (int off = 128; off > 0; off >>= 1) {
    if (tid < off) sv[tid] += sv[tid + off];
    __syncthreads();
  }
  float inv = 1.0f / fmaxf(sqrtf(sv[0]), EPSN);

  out[((size_t)b * n_slots + s) * D + tid] = v;          // unnormalized slot
  selnorm[(size_t)b * D + tid] = (_Float16)(v * inv);    // normalized, f16
  if (tid == 0) score[(size_t)b * N + idx] = 0.0f;       // never re-pick
}

// ---------------------------------------------------------------------------
// WMMA suppression pass (primary). One wave per 16-row tile.
// A (16x32 f16) = fnorm tile chunk, ISA A-layout:
//   lane L: row = L&15, halves 0..7 = K kb..kb+7, halves 8..15 = K kb+16..kb+23,
//   kb = (L>>4)*8, per 32-wide K chunk.
// B (32x16 f16) = sel_norm broadcast to all 16 columns; per-lane value depends
//   only on K: halves 0..7 = sel[(c*32) + (L>>4)*8 + 0..7], halves 8..15 = +16.
// C (16x16 f32): lane L, VGPR r -> row M = r + 8*(L>>4); every column equal.
// ---------------------------------------------------------------------------
__global__ __launch_bounds__(256)
void suppress_wmma(const _Float16* __restrict__ fn16,
                   const _Float16* __restrict__ selnorm,  // [B, D]
                   float* __restrict__ score,             // [B, N]
                   long totalTiles, int tilesPerBatch, int N, int D) {
  long wid = (long)((blockIdx.x * blockDim.x + threadIdx.x) >> 5);  // tile id
  if (wid >= totalTiles) return;   // whole-wave uniform exit: EXEC stays full
  const int lane = threadIdx.x & 31;

  const int b    = (int)(wid / tilesPerBatch);
  const int row0 = (int)(wid % tilesPerBatch) * 16;

  // Preload B operand: 8 chunks of the broadcast selected vector (f16).
  const _Float16* sn = selnorm + (size_t)b * D + ((lane >> 4) * 8);
  v16h bmat[8];
#pragma unroll
  for (int c = 0; c < 8; ++c) {
    v8h lo = *(const v8h*)(sn + c * 32);
    v8h hi = *(const v8h*)(sn + c * 32 + 16);
    bmat[c] = mk16(lo, hi);
  }

  // A operand base for this lane's row.
  const int r  = lane & 15;
  const int kb = (lane >> 4) * 8;
  const _Float16* ap = fn16 + ((size_t)b * N + row0 + r) * D + kb;

  v8f acc = {};
#pragma unroll
  for (int c = 0; c < 8; ++c) {          // 8 x K=32 covers D=256
    v8h lo = *(const v8h*)(ap + c * 32);
    v8h hi = *(const v8h*)(ap + c * 32 + 16);
    v16h a = mk16(lo, hi);
    acc = __builtin_amdgcn_wmma_f32_16x16x32_f16(
        /*neg_a=*/false, a, /*neg_b=*/false, bmat[c],
        /*c_mod=*/(short)0, acc, /*reuse_a=*/false, /*reuse_b=*/false);
  }

  // Columns 0..7 write back 16 rows (lanes 0-7 -> rows 0-7, lanes 16-23 -> 8-15).
  const int col = lane & 15;
  if (col < 8) {
    float sim = acc[0];
#pragma unroll
    for (int i = 1; i < 8; ++i) sim = (col == i) ? acc[i] : sim;
    sim = fminf(fmaxf(sim, 0.0f), 1.0f);
    const int rr = ((lane >> 4) * 8) + col;
    const size_t gi = (size_t)b * N + row0 + rr;
    score[gi] *= (1.0f - sim);
  }
}

// ---------------------------------------------------------------------------
// Fallback suppression (no f16 cache): one wave per row, f32 dot.
// ---------------------------------------------------------------------------
__global__ __launch_bounds__(256)
void suppress_f32(const float* __restrict__ feat,
                  const float* __restrict__ invn,
                  const _Float16* __restrict__ selnorm,
                  float* __restrict__ score,
                  long rowsTotal, int N, int D) {
  long wave = (long)((blockIdx.x * blockDim.x + threadIdx.x) >> 5);
  int lane = threadIdx.x & 31;
  if (wave >= rowsTotal) return;

  const int b = (int)(wave / N);
  const float* rp = feat + (size_t)wave * D + lane * 8;
  const _Float16* sn = selnorm + (size_t)b * D + lane * 8;
  float dot = 0.0f;
#pragma unroll
  for (int j = 0; j < 8; ++j) dot += rp[j] * (float)sn[j];
#pragma unroll
  for (int off = 16; off > 0; off >>= 1) dot += __shfl_xor(dot, off, 32);

  if (lane == 0) {
    float sim = dot * invn[wave];
    sim = fminf(fmaxf(sim, 0.0f), 1.0f);
    score[wave] *= (1.0f - sim);
  }
}

// ---------------------------------------------------------------------------
extern "C" void kernel_launch(void* const* d_in, const int* in_sizes, int n_in,
                              void* d_out, int out_size, void* d_ws, size_t ws_size,
                              hipStream_t stream) {
  const float* feat = (const float*)d_in[1];  // features [B, N, D] f32
  float* out = (float*)d_out;

  const int B = 32;                 // fixed by reference setup
  const int D = 256;
  const long total = (long)in_sizes[1];
  const int  N = (int)(total / ((long)B * D));        // 16384
  const int  n_slots = out_size / (B * D);            // 8
  const long rows = (long)B * N;

  // Workspace layout
  float* score  = (float*)d_ws;                        // rows f32
  float* invn   = score + rows;                        // rows f32
  _Float16* selnorm = (_Float16*)(invn + rows);        // B*D f16
  _Float16* fn16 = (_Float16*)((char*)(selnorm) + ((size_t)B * D * 2 + 63 & ~(size_t)63));
  size_t need = (size_t)rows * 8                       // score + invn
              + ((size_t)B * D * 2 + 64)               // selnorm (+pad)
              + (size_t)rows * D * 2;                  // fnorm f16 cache
  const bool useF16 = (ws_size >= need);

  // Pass 0: norms + (optional) f16 normalized cache. One wave per row.
  {
    long blocks = (rows + 7) / 8;   // 8 waves / 256-thread block
    init_norms<<<(unsigned)blocks, 256, 0, stream>>>(
        feat, score, invn, useF16 ? fn16 : nullptr, rows, D);
  }

  const long totalTiles = rows / 16;
  const int  tilesPerBatch = N / 16;

  for (int s = 0; s < n_slots; ++s) {
    select_slot<<<B, 256, 0, stream>>>(feat, score, selnorm, out, s, N, D, n_slots);
    if (useF16) {
      long blocks = (totalTiles + 7) / 8;   // 8 tiles (waves) per block
      suppress_wmma<<<(unsigned)blocks, 256, 0, stream>>>(
          fn16, selnorm, score, totalTiles, tilesPerBatch, N, D);
    } else {
      long blocks = (rows + 7) / 8;
      suppress_f32<<<(unsigned)blocks, 256, 0, stream>>>(
          feat, invn, selnorm, score, rows, N, D);
    }
  }
}
